// DINOPQGOCLS_34437047779986
// MI455X (gfx1250) — matrix operand order repaired and checked by reading, not compiled, not run on hardware
//
#include <hip/hip_runtime.h>
#include <math.h>

typedef _Float16 v16h __attribute__((ext_vector_type(16)));
typedef _Float16 v8h  __attribute__((ext_vector_type(8)));
typedef float    v8f  __attribute__((ext_vector_type(8)));

// Problem constants
#define NPQ   4
#define KCB   2048
#define PQD   128
#define BATCH 32
#define HWSZ  784      // 28*28
#define DTOT  512
#define NTOT  25088    // BATCH*HWSZ

// d_out element offsets (all float32): z_q_full, loss, prob, idx
static constexpr size_t ZQ_ELEMS  = (size_t)BATCH * DTOT * HWSZ;          // 12,845,056
static constexpr size_t LOSS_OFF  = ZQ_ELEMS;                              // 12,845,056
static constexpr size_t PROB_OFF  = LOSS_OFF + 1;                          // 12,845,057
static constexpr size_t IDX_OFF   = PROB_OFF + (size_t)NPQ * NTOT * KCB;   // 218,365,953

// workspace byte offsets (all 256B aligned)
static constexpr size_t WS_CBN  = 0;                                  // f16: 4*2048*128
static constexpr size_t WS_ZN   = 2097152;                            // f16: 4*25088*128
static constexpr size_t WS_CSQ  = 27787264;                           // f32: 4*2048
static constexpr size_t WS_ZSQ  = 27820032;                           // f32: 4*25088
static constexpr size_t WS_INV  = 28221440;                           // f32: 4*25088
static constexpr size_t WS_IDXI = 28622848;                           // i32: 4*25088

static constexpr float LOSS_SCALE = 1.25f / ((float)NPQ * (float)NTOT * (float)PQD);

// ---------------------------------------------------------------------------
__global__ void init_loss_kernel(float* out) { out[LOSS_OFF] = 0.0f; }

// ---------------------------------------------------------------------------
// Normalize codebooks -> f16 row-major [q][k][d], plus csq = sum(c_norm^2)
__global__ __launch_bounds__(128)
void prep_cb_kernel(const float* __restrict__ cb, _Float16* __restrict__ cbN,
                    float* __restrict__ csq) {
  const int q = blockIdx.y, k = blockIdx.x, t = threadIdx.x;
  __shared__ float red[128];
  const size_t base = ((size_t)q * KCB + k) * PQD;
  const float v = cb[base + t];
  red[t] = v * v;
  __syncthreads();
  for (int s = 64; s > 0; s >>= 1) { if (t < s) red[t] += red[t + s]; __syncthreads(); }
  const float ss  = red[0];
  const float inv = 1.0f / fmaxf(sqrtf(ss), 1e-12f);
  cbN[base + t] = (_Float16)(v * inv);
  if (t == 0) csq[q * KCB + k] = ss * inv * inv;
}

// ---------------------------------------------------------------------------
// Normalize z rows -> f16 row-major [q][i][d], plus zsq and invnorm
__global__ __launch_bounds__(128)
void prep_z_kernel(const float* __restrict__ z, _Float16* __restrict__ zN,
                   float* __restrict__ zsq, float* __restrict__ invn) {
  const int q = blockIdx.y, i = blockIdx.x, t = threadIdx.x;
  const int b = i / HWSZ, hw = i % HWSZ;
  __shared__ float red[128];
  const float v = z[((size_t)b * DTOT + q * PQD + t) * HWSZ + hw];
  red[t] = v * v;
  __syncthreads();
  for (int s = 64; s > 0; s >>= 1) { if (t < s) red[t] += red[t + s]; __syncthreads(); }
  const float ss  = red[0];
  const float inv = 1.0f / fmaxf(sqrtf(ss), 1e-12f);
  zN[((size_t)q * NTOT + i) * PQD + t] = (_Float16)(v * inv);
  if (t == 0) {
    zsq[q * NTOT + i]  = ss * inv * inv;
    invn[q * NTOT + i] = inv;
  }
}

// ---------------------------------------------------------------------------
// Fused WMMA GEMM (z_norm @ cb_norm^T) + distance + argmin + softmax + prob store.
// Block: 256 threads = 8 waves. Block covers 16 rows x 2048 cols.
// Wave w covers columns [w*256, w*256+256) as 16 WMMA N-tiles.
__global__ __launch_bounds__(256)
void vq_core_kernel(const _Float16* __restrict__ zN, const _Float16* __restrict__ cbN,
                    const float* __restrict__ zsq, const float* __restrict__ csq,
                    float* __restrict__ out, int* __restrict__ idxInt) {
  const int q    = blockIdx.y;
  const int row0 = blockIdx.x * 16;
  const int tid  = threadIdx.x;
  const int wave = tid >> 5;
  const int lane = tid & 31;
  const int hsel = lane >> 4;       // 0 = lanes 0-15, 1 = lanes 16-31
  const int l16  = lane & 15;
  const int colBase = wave * 256;

  __shared__ float sZ[16];
  __shared__ float sPmin[128];
  __shared__ int   sPidx[128];
  __shared__ float sPsum[128];
  __shared__ float sFmin[16];
  __shared__ int   sFidx[16];
  __shared__ float sFinv[16];

  if (tid < 16) sZ[tid] = zsq[q * NTOT + row0 + tid];
  __syncthreads();

  const _Float16* aRow  = zN + ((size_t)q * NTOT + row0 + l16) * PQD;
  const _Float16* bBase = cbN + (size_t)q * KCB * PQD;

  v8f acc[16];
#pragma unroll
  for (int t = 0; t < 16; ++t) acc[t] = (v8f)0.0f;

  // K loop: 128 in steps of 32
#pragma unroll
  for (int ks = 0; ks < PQD; ks += 32) {
    // A: lane<16 -> row l16, K {ks..ks+15}; lane>=16 -> K {ks+16..ks+31}
    const int kb = ks + hsel * 16;
    v8h alo = *(const v8h*)(aRow + kb);
    v8h ahi = *(const v8h*)(aRow + kb + 8);
    v16h a = __builtin_shufflevector(alo, ahi, 0,1,2,3,4,5,6,7,8,9,10,11,12,13,14,15);
    // B: lane holds column colBase+t*16+l16, K {kk..kk+15}, kk = ks + hsel*16
    const int kk = ks + hsel * 16;
#pragma unroll
    for (int t = 0; t < 16; ++t) {
      const int col = colBase + t * 16 + l16;
      v16h b = *(const v16h*)(bBase + (size_t)col * PQD + kk);
      acc[t] = __builtin_amdgcn_wmma_f32_16x16x32_f16(
          false, a, false, b, (short)0, acc[t], false, false);
    }
  }

  // C layout: VGPR r, lane -> row = r + hsel*8, col = colBase + t*16 + l16
  float zs[8];
#pragma unroll
  for (int r = 0; r < 8; ++r) zs[r] = sZ[r + hsel * 8];

  // Pass 1: d = zsq + csq - 2S (overwrite acc), track per-row min + argmin
  float rmin[8]; int ridx[8];
#pragma unroll
  for (int r = 0; r < 8; ++r) { rmin[r] = 3.4e38f; ridx[r] = 0; }
  const float* csqQ = csq + q * KCB;
#pragma unroll
  for (int t = 0; t < 16; ++t) {
    const int col = colBase + t * 16 + l16;
    const float cs = csqQ[col];
#pragma unroll
    for (int r = 0; r < 8; ++r) {
      const float d = zs[r] + cs - 2.0f * acc[t][r];
      acc[t][r] = d;
      if (d < rmin[r] || (d == rmin[r] && col < ridx[r])) { rmin[r] = d; ridx[r] = col; }
    }
  }
  // reduce across the 16 lanes of each half
#pragma unroll
  for (int m = 1; m < 16; m <<= 1) {
#pragma unroll
    for (int r = 0; r < 8; ++r) {
      const float ov = __shfl_xor(rmin[r], m, 32);
      const int   oi = __shfl_xor(ridx[r], m, 32);
      if (ov < rmin[r] || (ov == rmin[r] && oi < ridx[r])) { rmin[r] = ov; ridx[r] = oi; }
    }
  }
  if (l16 == 0) {
#pragma unroll
    for (int r = 0; r < 8; ++r) {
      sPmin[wave * 16 + r + hsel * 8] = rmin[r];
      sPidx[wave * 16 + r + hsel * 8] = ridx[r];
    }
  }
  __syncthreads();
  if (tid < 16) {
    float mv = sPmin[tid]; int bi = sPidx[tid];
    for (int w = 1; w < 8; ++w) {
      const float v = sPmin[w * 16 + tid]; const int vi = sPidx[w * 16 + tid];
      if (v < mv || (v == mv && vi < bi)) { mv = v; bi = vi; }
    }
    sFmin[tid] = mv; sFidx[tid] = bi;
  }
  __syncthreads();

  // Pass 2: e = exp(min - d) (overwrite acc), row sums
  float mm[8];
#pragma unroll
  for (int r = 0; r < 8; ++r) mm[r] = sFmin[r + hsel * 8];
  float rsum[8];
#pragma unroll
  for (int r = 0; r < 8; ++r) rsum[r] = 0.0f;
#pragma unroll
  for (int t = 0; t < 16; ++t) {
#pragma unroll
    for (int r = 0; r < 8; ++r) {
      const float e = __expf(mm[r] - acc[t][r]);
      acc[t][r] = e;
      rsum[r] += e;
    }
  }
#pragma unroll
  for (int m = 1; m < 16; m <<= 1) {
#pragma unroll
    for (int r = 0; r < 8; ++r) rsum[r] += __shfl_xor(rsum[r], m, 32);
  }
  if (l16 == 0) {
#pragma unroll
    for (int r = 0; r < 8; ++r) sPsum[wave * 16 + r + hsel * 8] = rsum[r];
  }
  __syncthreads();
  if (tid < 16) {
    float s = 0.0f;
    for (int w = 0; w < 8; ++w) s += sPsum[w * 16 + tid];
    sFinv[tid] = 1.0f / s;
  }
  __syncthreads();

  float iv[8];
#pragma unroll
  for (int r = 0; r < 8; ++r) iv[r] = sFinv[r + hsel * 8];

  // Pass 3: stream prob to HBM (dominant cost; written exactly once)
  float* probOut = out + PROB_OFF;
#pragma unroll
  for (int t = 0; t < 16; ++t) {
    const int col = colBase + t * 16 + l16;
#pragma unroll
    for (int r = 0; r < 8; ++r) {
      const int rowg = row0 + r + hsel * 8;
      probOut[((size_t)q * NTOT + rowg) * KCB + col] = acc[t][r] * iv[r];
    }
  }
  if (tid < 16) {
    out[IDX_OFF + (size_t)q * NTOT + row0 + tid] = (float)sFidx[tid];
    idxInt[q * NTOT + row0 + tid] = sFidx[tid];
  }
}

// ---------------------------------------------------------------------------
// z_q_full = cb[idx] scattered back to (B,D,H,W); loss = 1.25*mean((z_q - z_norm)^2)
__global__ __launch_bounds__(128)
void scatter_loss_kernel(const float* __restrict__ z, const float* __restrict__ cb,
                         const int* __restrict__ idxInt, const float* __restrict__ invn,
                         float* __restrict__ out) {
  const int q = blockIdx.y, i = blockIdx.x, t = threadIdx.x;
  const int b = i / HWSZ, hw = i % HWSZ;
  const int ci = idxInt[q * NTOT + i];
  const float c = cb[((size_t)q * KCB + ci) * PQD + t];
  const size_t zoff = ((size_t)b * DTOT + q * PQD + t) * HWSZ + hw;
  out[zoff] = c;  // straight-through forward value
  const float zn = z[zoff] * invn[q * NTOT + i];
  const float df = c - zn;
  __shared__ float red[128];
  red[t] = df * df;
  __syncthreads();
  for (int s = 64; s > 0; s >>= 1) { if (t < s) red[t] += red[t + s]; __syncthreads(); }
  if (t == 0) atomicAdd(out + LOSS_OFF, red[0] * LOSS_SCALE);
}

// ---------------------------------------------------------------------------
extern "C" void kernel_launch(void* const* d_in, const int* in_sizes, int n_in,
                              void* d_out, int out_size, void* d_ws, size_t ws_size,
                              hipStream_t stream) {
  (void)in_sizes; (void)n_in; (void)out_size; (void)ws_size;
  const float* z  = (const float*)d_in[0];
  const float* cb = (const float*)d_in[1];
  float* out = (float*)d_out;
  char* ws = (char*)d_ws;

  _Float16* cbN = (_Float16*)(ws + WS_CBN);
  _Float16* zN  = (_Float16*)(ws + WS_ZN);
  float* csq  = (float*)(ws + WS_CSQ);
  float* zsq  = (float*)(ws + WS_ZSQ);
  float* invn = (float*)(ws + WS_INV);
  int*   idxI = (int*)(ws + WS_IDXI);

  init_loss_kernel<<<dim3(1), dim3(1), 0, stream>>>(out);
  prep_cb_kernel<<<dim3(KCB, NPQ), dim3(128), 0, stream>>>(cb, cbN, csq);
  prep_z_kernel<<<dim3(NTOT, NPQ), dim3(128), 0, stream>>>(z, zN, zsq, invn);
  vq_core_kernel<<<dim3(NTOT / 16, NPQ), dim3(256), 0, stream>>>(zN, cbN, zsq, csq, out, idxI);
  scatter_loss_kernel<<<dim3(NTOT, NPQ), dim3(128), 0, stream>>>(z, cb, idxI, invn, out);
}
